// GCN_Module_71494025609306
// MI455X (gfx1250) — matrix operand order repaired and checked by reading, not compiled, not run on hardware
//
#include <hip/hip_runtime.h>

// ---------------------------------------------------------------------------
// GCN relation module for MI455X (gfx1250, wave32, WMMA).
// Pipeline per sub-graph i (12x):
//   gemm(theta) -> gemm(phi) -> fused sim+mask+softmax (LDS-resident rows)
//   -> gemm(P @ x) -> gemm(.. @ Wgcn, +LN partial stats) -> LN stats reduce
//   -> LN apply + relu + accumulate into out.
// All GEMMs: bf16 inputs, v_wmma_f32_16x16x32_bf16, fp32 accumulate.
// LDS tiles are filled with global_load_async_to_lds_b128 (ASYNCcnt path).
// ---------------------------------------------------------------------------

typedef __attribute__((ext_vector_type(16))) __bf16 v16bf;
typedef __attribute__((ext_vector_type(8)))  float  v8f;
typedef __attribute__((ext_vector_type(4)))  float  f32x4;

union FragU { f32x4 f4[2]; v16bf v; };

#define DEVFN static __device__ __forceinline__

constexpr int Bb  = 8;
constexpr int Sd  = 1280;
constexpr int Cd  = 512;
constexpr int NGc = 12;

// ---- bf16 helpers (storage is raw ushort bits; round-to-nearest-even) ------
DEVFN unsigned short f2bf(float f) {
  unsigned u = __float_as_uint(f);
  u += 0x7FFFu + ((u >> 16) & 1u);
  return (unsigned short)(u >> 16);
}

// ---- async global -> LDS 16B copy (gfx1250, tracked with ASYNCcnt) ---------
// Generic pointers to __shared__ carry the LDS byte offset in their low 32
// bits, which is exactly what the instruction's VDST (LDS address) expects.
DEVFN void async_copy_b128(void* lds_dst, const void* gsrc) {
  unsigned lds = (unsigned)(unsigned long long)lds_dst;
  asm volatile("global_load_async_to_lds_b128 %0, %1, off"
               :: "v"(lds), "v"(gsrc) : "memory");
}
DEVFN void wait_async0() {
  asm volatile("s_wait_asynccnt 0x0" ::: "memory");
}

// ---- WMMA fragment loads from LDS tiles (row-major [row][K], K-contig) -----
// A matrix 16x32 bf16 (ISA 7.12.2): lane m=L%16; lanes<16: K 0..7 & 16..23,
// lanes>=16: K 8..15 & 24..31  => two 16B chunks at off and off+16.
DEVFN v16bf load_frag_a(const unsigned short* base, int stride) {
  int lane = threadIdx.x & 31;
  const unsigned short* p = base + (lane & 15) * stride + ((lane >> 4) << 3);
  FragU u;
  u.f4[0] = *(const f32x4*)(p);
  u.f4[1] = *(const f32x4*)(p + 16);
  return u.v;
}
// B matrix 32x16 bf16: lane n=L%16; lanes<16: K 0..7 & 16..23,
// lanes>=16: K 8..15 & 24..31 => two 16B chunks at off and off+16 from Bt[n][k].
DEVFN v16bf load_frag_b(const unsigned short* base, int stride) {
  int lane = threadIdx.x & 31;
  const unsigned short* p = base + (lane & 15) * stride + ((lane >> 4) << 3);
  FragU u;
  u.f4[0] = *(const f32x4*)(p);
  u.f4[1] = *(const f32x4*)(p + 16);
  return u.v;
}

DEVFN v8f wmma_bf16(v16bf a, v16bf b, v8f c) {
  return __builtin_amdgcn_wmma_f32_16x16x32_bf16(
      false, a, false, b, (short)0, c, false, false);
}

// ---------------------------------------------------------------------------
// Generic bf16 GEMM: C[M,N] = A[M,K] * Bt[N,K]^T (+bias[n])
// 128x128 block tile, 256 threads (8 waves), K-step 32.
// Optional bf16/f32 outputs; optional per-block (sum,sumsq) partials for LN.
// ---------------------------------------------------------------------------
__global__ __launch_bounds__(256)
void gemm_bf16(const unsigned short* __restrict__ A, long strideAz, int lda,
               const unsigned short* __restrict__ Bt, long strideBz, int ldb,
               const float* __restrict__ bias,
               unsigned short* __restrict__ outB, float* __restrict__ outF,
               long strideCz, int ldc, int K,
               float* __restrict__ partials, int mblocks_per_b) {
  __shared__ unsigned short As[128 * 40];   // stride 40 (=20 dwords) -> no bank conflicts
  __shared__ unsigned short Bs[128 * 40];
  __shared__ float red[512];

  int z = blockIdx.z;
  int mb = blockIdx.x, nb = blockIdx.y;
  int tid = threadIdx.x, lane = tid & 31, wid = tid >> 5;
  int wm = wid >> 2, wn = wid & 3;          // 2x4 wave grid over 128x128 tile

  const unsigned short* Ag = A + z * strideAz + (long)mb * 128 * lda;
  const unsigned short* Bg = Bt + z * strideBz + (long)nb * 128 * ldb;
  long cOff = z * strideCz;

  v8f acc[4][2];
#pragma unroll
  for (int i = 0; i < 4; ++i)
#pragma unroll
    for (int j = 0; j < 2; ++j)
#pragma unroll
      for (int r = 0; r < 8; ++r) acc[i][j][r] = 0.0f;

  for (int k0 = 0; k0 < K; k0 += 32) {
    __syncthreads();
    // stage 128x32 bf16 tiles of A and Bt via async global->LDS b128 copies
#pragma unroll
    for (int c = tid; c < 512; c += 256) {
      int r = c >> 2, kc = c & 3;
      async_copy_b128(As + r * 40 + kc * 8, Ag + (long)r * lda + k0 + kc * 8);
      async_copy_b128(Bs + r * 40 + kc * 8, Bg + (long)r * ldb + k0 + kc * 8);
    }
    if (k0 + 32 < K) {                       // global_prefetch_b8 for next slab
      int r = tid >> 1, kc = tid & 1;
      __builtin_prefetch(Ag + (long)r * lda + k0 + 32 + kc * 16, 0, 3);
      __builtin_prefetch(Bg + (long)r * ldb + k0 + 32 + kc * 16, 0, 3);
    }
    wait_async0();
    __syncthreads();

    v16bf afrag[4], bfrag[2];
#pragma unroll
    for (int j = 0; j < 2; ++j) bfrag[j] = load_frag_b(Bs + (wn * 32 + j * 16) * 40, 40);
#pragma unroll
    for (int i = 0; i < 4; ++i) afrag[i] = load_frag_a(As + (wm * 64 + i * 16) * 40, 40);
#pragma unroll
    for (int i = 0; i < 4; ++i)
#pragma unroll
      for (int j = 0; j < 2; ++j) acc[i][j] = wmma_bf16(afrag[i], bfrag[j], acc[i][j]);
  }

  // epilogue: C/D layout -> lane n = L%16, row = vgpr + 8*(L>=16)
  float psum = 0.f, psq = 0.f;
  int rbase = (lane >> 4) << 3;
  int cbase = lane & 15;
#pragma unroll
  for (int i = 0; i < 4; ++i) {
#pragma unroll
    for (int j = 0; j < 2; ++j) {
      int col = nb * 128 + wn * 32 + j * 16 + cbase;
      float bv = bias ? bias[col] : 0.f;
#pragma unroll
      for (int r = 0; r < 8; ++r) {
        int row = mb * 128 + wm * 64 + i * 16 + rbase + r;
        float v = acc[i][j][r] + bv;
        long o = cOff + (long)row * ldc + col;
        if (outB) outB[o] = f2bf(v);
        if (outF) outF[o] = v;
        psum += v; psq += v * v;
      }
    }
  }

  if (partials) {                           // per-block LN partial sums
    red[tid] = psum; red[256 + tid] = psq;
    __syncthreads();
    for (int k = 128; k > 0; k >>= 1) {
      if (tid < k) { red[tid] += red[tid + k]; red[256 + tid] += red[256 + tid + k]; }
      __syncthreads();
    }
    if (tid == 0) {
      int bb = (mb * 128) / Sd;
      int slot = (mb % mblocks_per_b) * gridDim.y + nb;
      float* p = partials + ((long)bb * mblocks_per_b * gridDim.y + slot) * 2;
      p[0] = red[0]; p[1] = red[256];
    }
  }
}

// ---------------------------------------------------------------------------
// Fused sim = theta @ phi^T * inv_sqrt, position mask, row softmax -> P(bf16)
// One block = 32 rows of one batch; full 32x1280 fp32 logits live in LDS
// (~203 KB dynamic LDS, a CDNA5 320KB/WGP capability).
// ---------------------------------------------------------------------------
__global__ __launch_bounds__(256)
void attn_softmax(const unsigned short* __restrict__ theta,
                  const unsigned short* __restrict__ phi,
                  const unsigned char* __restrict__ mask,
                  unsigned short* __restrict__ P,
                  float* __restrict__ relOut) {
  extern __shared__ char smem[];
  float* simbuf = (float*)smem;                               // 32*1280 f32
  unsigned short* thS = (unsigned short*)(simbuf + 32 * Sd);  // 32*520 bf16
  unsigned short* phS = thS + 32 * 520;                       // 128*40 bf16
  float* rowmax = (float*)(phS + 128 * 40);                   // 32
  float* rowsum = rowmax + 32;                                // 32

  int b = blockIdx.y, rb = blockIdx.x, row0 = rb * 32;
  int tid = threadIdx.x, lane = tid & 31, wid = tid >> 5;
  const float inv_sqrt = 0.04419417382415922f;   // 1/sqrt(512)

  // stage theta strip 32x512 once (stride 520 avoids bank conflicts)
  const unsigned short* thG = theta + ((long)b * Sd + row0) * Cd;
  for (int c = tid; c < 32 * (Cd / 8); c += 256) {
    int r = c >> 6, kc = c & 63;
    async_copy_b128(thS + r * 520 + kc * 8, thG + (long)r * Cd + kc * 8);
  }

  int mt = wid >> 2;               // wave's 16-row tile (0..1)
  int ln0 = (wid & 3) * 2;         // wave's pair of 16-col tiles within 128

  for (int cb = 0; cb < Sd / 128; ++cb) {
    v8f acc0, acc1;
#pragma unroll
    for (int r = 0; r < 8; ++r) { acc0[r] = 0.f; acc1[r] = 0.f; }
    const unsigned short* phG = phi + ((long)b * Sd + cb * 128) * Cd;
    for (int k0 = 0; k0 < Cd; k0 += 32) {
      __syncthreads();
      for (int c = tid; c < 128 * 4; c += 256) {     // stage phi 128x32 chunk
        int r = c >> 2, kc = c & 3;
        async_copy_b128(phS + r * 40 + kc * 8, phG + (long)r * Cd + k0 + kc * 8);
      }
      wait_async0();
      __syncthreads();
      v16bf a  = load_frag_a(thS + mt * 16 * 520 + k0, 520);
      v16bf b0 = load_frag_b(phS + (ln0)     * 16 * 40, 40);
      v16bf b1 = load_frag_b(phS + (ln0 + 1) * 16 * 40, 40);
      acc0 = wmma_bf16(a, b0, acc0);
      acc1 = wmma_bf16(a, b1, acc1);
    }
    int rr = mt * 16 + ((lane >> 4) << 3);
    int cc = cb * 128 + ln0 * 16 + (lane & 15);
#pragma unroll
    for (int r = 0; r < 8; ++r) {
      simbuf[(rr + r) * Sd + cc]      = acc0[r] * inv_sqrt;
      simbuf[(rr + r) * Sd + cc + 16] = acc1[r] * inv_sqrt;
    }
  }
  __syncthreads();

  // mask + row max + row sum  (8 threads per row, shuffle reduce in-wave)
  {
    int g = tid & 7, r = tid >> 3;
    const unsigned char* mrow = mask + ((long)b * Sd + row0 + r) * Sd;
    float mx = -__builtin_inff();
    for (int c = g; c < Sd; c += 8) {
      float v = simbuf[r * Sd + c];
      if (mrow[c]) v = -__builtin_inff();
      simbuf[r * Sd + c] = v;
      mx = fmaxf(mx, v);
    }
    mx = fmaxf(mx, __shfl_xor(mx, 1, 32));
    mx = fmaxf(mx, __shfl_xor(mx, 2, 32));
    mx = fmaxf(mx, __shfl_xor(mx, 4, 32));
    float s = 0.f;
    for (int c = g; c < Sd; c += 8) s += __expf(simbuf[r * Sd + c] - mx);
    s += __shfl_xor(s, 1, 32);
    s += __shfl_xor(s, 2, 32);
    s += __shfl_xor(s, 4, 32);
    if (g == 0) { rowmax[r] = mx; rowsum[r] = 1.0f / s; }
  }
  __syncthreads();

  // write normalized P (bf16) and, on last sub-graph, fp32 relation_graph
  // (relation stream is write-once -> non-temporal, keep L2 for reused tiles)
  for (int r = 0; r < 32; ++r) {
    float mx = rowmax[r], is = rowsum[r];
    long orow = ((long)b * Sd + row0 + r) * Sd;
    for (int c = tid; c < Sd; c += 256) {
      float v = __expf(simbuf[r * Sd + c] - mx) * is;
      P[orow + c] = f2bf(v);
      if (relOut) __builtin_nontemporal_store(v, &relOut[orow + c]);
    }
  }
}

// ---------------------------------------------------------------------------
// small kernels
// ---------------------------------------------------------------------------
__global__ void zero_f32(float* p, int n) {
  int i = blockIdx.x * 256 + threadIdx.x;
  if (i < n) p[i] = 0.f;
}

__global__ void pos_kernel(const float* __restrict__ box, float* __restrict__ pos, int n) {
  int i = blockIdx.x * 256 + threadIdx.x;
  if (i < n) {
    pos[i * 2]     = (box[i * 4]     + box[i * 4 + 2]) * 0.5f;
    pos[i * 2 + 1] = (box[i * 4 + 1] + box[i * 4 + 3]) * 0.5f;
  }
}

__global__ void mask_kernel(const float* __restrict__ pos, const int* __restrict__ ow,
                            unsigned char* __restrict__ mask) {
  int b = blockIdx.y, n = blockIdx.x;
  float t = 0.2f * (float)ow[0];
  float t2 = t * t;                              // dist > t  <=>  d2 > t^2
  long rowi = (long)b * Sd + n;
  float cx = pos[rowi * 2], cy = pos[rowi * 2 + 1];
  const float* pb = pos + (long)b * Sd * 2;
  unsigned char* mrow = mask + rowi * Sd;
  for (int m = threadIdx.x; m < Sd; m += 256) {
    float dx = cx - pb[m * 2], dy = cy - pb[m * 2 + 1];
    mrow[m] = (dx * dx + dy * dy) > t2 ? 1 : 0;
  }
}

// fp32 -> bf16 transpose (outT[c][r]) + optional straight bf16 copy (outS[r][c])
__global__ void tconv(const float* __restrict__ in, unsigned short* __restrict__ outT,
                      unsigned short* __restrict__ outS, int R, int Cc,
                      long inZ, long outTZ, long outSZ) {
  __shared__ float tile[32][33];
  int z = blockIdx.z;
  in += (long)z * inZ;
  outT += (long)z * outTZ;
  if (outS) outS += (long)z * outSZ;
  int c0 = blockIdx.x * 32, r0 = blockIdx.y * 32;
  int tx = threadIdx.x & 31, ty = threadIdx.x >> 5;
#pragma unroll
  for (int j = 0; j < 32; j += 8) {
    float v = in[(long)(r0 + ty + j) * Cc + c0 + tx];
    tile[ty + j][tx] = v;
    if (outS) outS[(long)(r0 + ty + j) * Cc + c0 + tx] = f2bf(v);
  }
  __syncthreads();
#pragma unroll
  for (int j = 0; j < 32; j += 8)
    outT[(long)(c0 + ty + j) * R + r0 + tx] = f2bf(tile[tx][ty + j]);
}

__global__ void ln_stats(const float* __restrict__ partials, float* __restrict__ muvar) {
  int b = blockIdx.x, t = threadIdx.x;       // 64 threads, 40 partial slots
  __shared__ float s1[64], s2[64];
  float a = 0.f, q = 0.f;
  if (t < 40) { a = partials[((long)b * 40 + t) * 2]; q = partials[((long)b * 40 + t) * 2 + 1]; }
  s1[t] = a; s2[t] = q;
  __syncthreads();
  for (int k = 32; k > 0; k >>= 1) {
    if (t < k) { s1[t] += s1[t + k]; s2[t] += s2[t + k]; }
    __syncthreads();
  }
  if (t == 0) {
    float n = (float)Sd * (float)Cd;
    float mu = s1[0] / n;
    float var = s2[0] / n - mu * mu;
    muvar[b * 2] = mu;
    muvar[b * 2 + 1] = rsqrtf(var + 1e-5f);
  }
}

__global__ void ln_apply(const float* __restrict__ agg, const float* __restrict__ muvar,
                         const float* __restrict__ lns, const float* __restrict__ lnb,
                         float* __restrict__ out) {
  int idx = blockIdx.x * 256 + threadIdx.x;          // BSC = 5,242,880 exact
  const int SCc = Sd * Cd;
  int b = idx / SCc;
  int r = idx - b * SCc;
  float mu = muvar[b * 2], rstd = muvar[b * 2 + 1];
  float v = (agg[idx] - mu) * rstd * lns[r] + lnb[r];
  out[idx] += fmaxf(v, 0.f);
}

// ---------------------------------------------------------------------------
extern "C" void kernel_launch(void* const* d_in, const int* in_sizes, int n_in,
                              void* d_out, int out_size, void* d_ws, size_t ws_size,
                              hipStream_t stream) {
  const float* x   = (const float*)d_in[0];
  const float* box = (const float*)d_in[1];
  const float* Wth = (const float*)d_in[2];
  const float* bth = (const float*)d_in[3];
  const float* Wph = (const float*)d_in[4];
  const float* bph = (const float*)d_in[5];
  const float* Wg  = (const float*)d_in[6];
  const float* lns = (const float*)d_in[7];
  const float* lnb = (const float*)d_in[8];
  const int*   OWp = (const int*)d_in[10];

  constexpr long SC  = (long)Sd * Cd;        // 655360
  constexpr long BSC = (long)Bb * SC;        // 5,242,880
  constexpr long BSS = (long)Bb * Sd * Sd;   // 13,107,200
  constexpr long WSZ = (long)Cd * Cd;        // 262144

  float* out = (float*)d_out;                // [B,S,C]
  float* rel = out + BSC;                    // [B,S,S] relation_graph (last i)

  char* w = (char*)d_ws;
  long off = 0;
  auto alloc = [&](long bytes) {
    off = (off + 255) & ~255L;
    long o = off; off += bytes; return o;
  };
  unsigned short* xbf   = (unsigned short*)(w + alloc(BSC * 2));
  unsigned short* xT    = (unsigned short*)(w + alloc(BSC * 2));       // [b][C][S]
  unsigned short* WthT  = (unsigned short*)(w + alloc(NGc * WSZ * 2)); // [i][N][K]
  unsigned short* WphT  = (unsigned short*)(w + alloc(NGc * WSZ * 2));
  unsigned short* WgT   = (unsigned short*)(w + alloc(NGc * WSZ * 2));
  unsigned short* theta = (unsigned short*)(w + alloc(BSC * 2));       // reused as agg0
  unsigned short* phi   = (unsigned short*)(w + alloc(BSC * 2));
  unsigned short* P     = (unsigned short*)(w + alloc(BSS * 2));
  float* aggf           = (float*)(w + alloc(BSC * 4));
  float* posb           = (float*)(w + alloc((long)Bb * Sd * 2 * 4));
  unsigned char* maskb  = (unsigned char*)(w + alloc(BSS));
  float* part           = (float*)(w + alloc(8 * 40 * 2 * 4));
  float* muvar          = (float*)(w + alloc(64));

  // ---- one-time prep ----
  zero_f32<<<(int)(BSC / 256), 256, 0, stream>>>(out, (int)BSC);
  pos_kernel<<<(Bb * Sd) / 256, 256, 0, stream>>>(box, posb, Bb * Sd);
  mask_kernel<<<dim3(Sd, Bb), 256, 0, stream>>>(posb, OWp, maskb);
  tconv<<<dim3(Cd / 32, Sd / 32, Bb), 256, 0, stream>>>(x, xT, xbf, Sd, Cd, SC, SC, SC);
  tconv<<<dim3(Cd / 32, Cd / 32, NGc), 256, 0, stream>>>(Wth, WthT, nullptr, Cd, Cd, WSZ, WSZ, 0);
  tconv<<<dim3(Cd / 32, Cd / 32, NGc), 256, 0, stream>>>(Wph, WphT, nullptr, Cd, Cd, WSZ, WSZ, 0);
  tconv<<<dim3(Cd / 32, Cd / 32, NGc), 256, 0, stream>>>(Wg,  WgT,  nullptr, Cd, Cd, WSZ, WSZ, 0);

  constexpr int SMEM = 32 * Sd * 4 + 32 * 520 * 2 + 128 * 40 * 2 + 64 * 4;  // 207,616 B
  (void)hipFuncSetAttribute((const void*)attn_softmax,
                            hipFuncAttributeMaxDynamicSharedMemorySize, SMEM);

  // ---- 12 sub-graphs ----
  for (int i = 0; i < NGc; ++i) {
    // theta = x @ Wth[i] + bth[i]   (bf16 out)
    gemm_bf16<<<dim3(80, 4, 1), 256, 0, stream>>>(
        xbf, 0, Cd, WthT + (long)i * WSZ, 0, Cd, bth + (long)i * Cd,
        theta, nullptr, 0, Cd, Cd, nullptr, 0);
    // phi = x @ Wph[i] + bph[i]
    gemm_bf16<<<dim3(80, 4, 1), 256, 0, stream>>>(
        xbf, 0, Cd, WphT + (long)i * WSZ, 0, Cd, bph + (long)i * Cd,
        phi, nullptr, 0, Cd, Cd, nullptr, 0);
    // P = softmax(mask(theta @ phi^T / sqrt(512)))   (+ fp32 relation on last i)
    attn_softmax<<<dim3(Sd / 32, Bb), 256, SMEM, stream>>>(
        theta, phi, maskb, P, (i == NGc - 1) ? rel : nullptr);
    // agg0 = P @ x   (per-batch, bf16 out, reuses theta buffer)
    gemm_bf16<<<dim3(Sd / 128, 4, Bb), 256, 0, stream>>>(
        P, (long)Sd * Sd, Sd, xT, (long)Cd * Sd, Sd, nullptr,
        theta, nullptr, SC, Cd, Sd, nullptr, 0);
    // agg = agg0 @ Wg[i]  (fp32 out + LN partial stats)
    gemm_bf16<<<dim3(80, 4, 1), 256, 0, stream>>>(
        theta, 0, Cd, WgT + (long)i * WSZ, 0, Cd, nullptr,
        nullptr, aggf, 0, Cd, Cd, part, 10);
    // LayerNorm over [S,C] per batch, then relu-accumulate
    ln_stats<<<Bb, 64, 0, stream>>>(part, muvar);
    ln_apply<<<(int)(BSC / 256), 256, 0, stream>>>(
        aggf, muvar, lns + (long)i * SC, lnb + (long)i * SC, out);
  }
}
